// SRCNN_36928128811191
// MI455X (gfx1250) — matrix operand order repaired and checked by reading, not compiled, not run on hardware
//
#include <hip/hip_runtime.h>
#include <hip/hip_bf16.h>

typedef __attribute__((ext_vector_type(16))) _Float16 v16h;
typedef __attribute__((ext_vector_type(8)))  _Float16 v8h;
typedef __attribute__((ext_vector_type(4)))  _Float16 v4h;
typedef __attribute__((ext_vector_type(8)))  float    v8f;
typedef __attribute__((ext_vector_type(4)))  float    v4f;

#define A_COEF (-0.75f)

__device__ __forceinline__ float cubic_w(float d) {
  d = fabsf(d);
  if (d <= 1.0f) return (A_COEF + 2.0f) * d * d * d - (A_COEF + 3.0f) * d * d + 1.0f;
  if (d < 2.0f)  return A_COEF * d * d * d - 5.0f * A_COEF * d * d + 8.0f * A_COEF * d - 4.0f * A_COEF;
  return 0.0f;
}

__device__ __forceinline__ v16h cat8(v8h lo, v8h hi) {
  return __builtin_shufflevector(lo, hi, 0, 1, 2, 3, 4, 5, 6, 7,
                                 8, 9, 10, 11, 12, 13, 14, 15);
}

// ---------------------------------------------------------------------------
// Kernel 1: bicubic x8 upscale, NHWC fp32 [16,32,32,4] -> f16 NHWC [16,256,256,4]
// ---------------------------------------------------------------------------
__global__ __launch_bounds__(256) void k_upsample(const float* __restrict__ x,
                                                  _Float16* __restrict__ up) {
  int o  = blockIdx.x * blockDim.x + threadIdx.x;  // 16*256*256 threads
  int b  = o >> 16;
  int oy = (o >> 8) & 255;
  int ox = o & 255;

  float sy = (oy + 0.5f) * 0.125f - 0.5f;
  float sx = (ox + 0.5f) * 0.125f - 0.5f;
  int by = (int)floorf(sy), bx = (int)floorf(sx);
  float ty = sy - (float)by, tx = sx - (float)bx;

  float wy[4], wx[4];
  int iy[4], ix[4];
#pragma unroll
  for (int i = 0; i < 4; ++i) {
    wy[i] = cubic_w(ty - (float)(i - 1));
    wx[i] = cubic_w(tx - (float)(i - 1));
    iy[i] = min(max(by + i - 1, 0), 31);
    ix[i] = min(max(bx + i - 1, 0), 31);
  }

  v4f acc = {};
#pragma unroll
  for (int i = 0; i < 4; ++i) {
#pragma unroll
    for (int j = 0; j < 4; ++j) {
      float wgt = wy[i] * wx[j];
      const v4f px = *(const v4f*)(x + (((b * 32 + iy[i]) * 32) + ix[j]) * 4);
      acc += wgt * px;
    }
  }
  v4h hv;
  hv[0] = (_Float16)acc[0]; hv[1] = (_Float16)acc[1];
  hv[2] = (_Float16)acc[2]; hv[3] = (_Float16)acc[3];
  *(v4h*)(up + (size_t)o * 4) = hv;
}

// ---------------------------------------------------------------------------
// Kernel 2: fused conv1 (9x9, 4->64, pad4) + ReLU + conv2 (1x1, 64->32) + ReLU
// Implicit GEMM via v_wmma_f32_16x16x32_f16.
//   M = 16 pixels, N = 64 (4 waves x 16), K = 4ch*81taps = 324 -> 11 chunks of 32.
// LDS layouts are fragment-friendly so per-lane gathers are ds_load_b128.
// ---------------------------------------------------------------------------
#define KCH 11
#define KTOT (KCH * 32)  // 352

__global__ __launch_bounds__(128) void k_conv12(const _Float16* __restrict__ up,
                                                const float* __restrict__ w1,
                                                const float* __restrict__ b1,
                                                const float* __restrict__ w2,
                                                const float* __restrict__ b2,
                                                _Float16* __restrict__ h2) {
  __shared__ _Float16 ldsB[64][KTOT];  // conv1 weights, fragment-major [n][k]
  __shared__ _Float16 ldsA[16][KTOT];  // im2col tile  [pixel][k]
  __shared__ _Float16 ldsH[16][72];    // conv1 activations, 16 px x 64 ch

  const int tid  = threadIdx.x;
  const int lane = tid & 31;
  const int wv   = tid >> 5;
  const int w0   = blockIdx.x * 16;
  const int h    = blockIdx.y;
  const int b    = blockIdx.z;

  // --- Stage conv1 weights fragment-major: ldsB[n][g*32 + ti*4 + c] ---
  {
    const int n    = tid >> 1;          // 0..63
    const int half = tid & 1;           // each half stages 176 k-slots
    for (int kt = half * 176; kt < half * 176 + 176; ++kt) {
      int g = kt >> 5, k = kt & 31;
      int t = g * 8 + (k >> 2), c = k & 3;
      float v = (t < 81) ? w1[(n * 4 + c) * 81 + t] : 0.0f;
      ldsB[n][kt] = (_Float16)v;
    }
  }
  // --- Stage im2col A with v4h (4 contiguous channels per tap) ---
  for (int e = tid; e < 16 * 88; e += 128) {
    int p = e / 88, t = e % 88;         // t == tap index (g*8 + ti)
    int g = t >> 3, ti = t & 7;
    v4h v = {};
    if (t < 81) {
      int dy = t / 9 - 4, dx = t % 9 - 4;
      int ih = h + dy, iw = w0 + p + dx;
      if (ih >= 0 && ih < 256 && iw >= 0 && iw < 256)
        v = *(const v4h*)(up + (((b * 256 + ih) * 256) + iw) * 4);
    }
    *(v4h*)&ldsA[p][g * 32 + ti * 4] = v;
  }
  __syncthreads();

  const int m   = lane & 15;   // A row (pixel), B/D column
  const int hi  = lane >> 4;
  const int kbA = hi * 16;     // A K-base within chunk (0 or 16)
  const int kbB = hi * 16;     // B K-base within chunk (0 or 16)
  const int nw  = wv * 16;     // this wave's N slice of conv1

  const _Float16* arow = &ldsA[m][0];
  const _Float16* brow = &ldsB[nw + m][0];

  v8f acc = {};
#pragma unroll
  for (int g = 0; g < KCH; ++g) {
    v8h alo = *(const v8h*)(arow + g * 32 + kbA);
    v8h ahi = *(const v8h*)(arow + g * 32 + kbA + 8);
    v8h blo = *(const v8h*)(brow + g * 32 + kbB);
    v8h bhi = *(const v8h*)(brow + g * 32 + kbB + 8);
    acc = __builtin_amdgcn_wmma_f32_16x16x32_f16(false, cat8(alo, ahi),
                                                 false, cat8(blo, bhi),
                                                 (short)0, acc, false, false);
  }

  // bias + ReLU -> f16 LDS tile (16 px x 64 ch)
  {
    int ch = nw + m;
    float bias = b1[ch];
#pragma unroll
    for (int r = 0; r < 8; ++r) {
      float f = fmaxf(acc[r] + bias, 0.0f);
      ldsH[r + hi * 8][ch] = (_Float16)f;
    }
  }
  __syncthreads();

  // conv2: 1x1 GEMM 16px x 32ch x K=64 (waves 0,1)
  if (wv < 2) {
    const int n2 = wv * 16 + m;  // conv2 out channel 0..31
    const _Float16* hrow = &ldsH[m][0];
    v8f acc2 = {};
#pragma unroll
    for (int kc = 0; kc < 2; ++kc) {
      v8h alo = *(const v8h*)(hrow + kc * 32 + kbA);
      v8h ahi = *(const v8h*)(hrow + kc * 32 + kbA + 8);
      v16h bb;
#pragma unroll
      for (int q = 0; q < 4; ++q) {
        v4f wq = *(const v4f*)(w2 + n2 * 64 + kc * 32 + kbB + q * 4);
        bb[q * 4 + 0] = (_Float16)wq[0];
        bb[q * 4 + 1] = (_Float16)wq[1];
        bb[q * 4 + 2] = (_Float16)wq[2];
        bb[q * 4 + 3] = (_Float16)wq[3];
      }
      acc2 = __builtin_amdgcn_wmma_f32_16x16x32_f16(false, cat8(alo, ahi),
                                                    false, bb,
                                                    (short)0, acc2, false, false);
    }
    float bias2 = b2[n2];
#pragma unroll
    for (int r = 0; r < 8; ++r) {
      float f = fmaxf(acc2[r] + bias2, 0.0f);
      int px = w0 + r + hi * 8;
      h2[(((b * 256 + h) * 256) + px) * 32 + n2] = (_Float16)f;
    }
  }
}

// ---------------------------------------------------------------------------
// Kernel 3: conv3 (5x5, 32->4, pad2). K = 25 taps x 32 ch = 25 WMMA chunks.
// w3 pre-transposed into fragment-major LDS once per block; A fragments are
// two b128 global loads per tap (channels contiguous in h2).
// ---------------------------------------------------------------------------
__global__ __launch_bounds__(64) void k_conv3(const _Float16* __restrict__ h2,
                                              const float* __restrict__ w3,
                                              const float* __restrict__ b3,
                                              float* __restrict__ out) {
  __shared__ _Float16 ldsW[16][800];  // [n][t*32 + k]; zero rows for n >= 4

  const int tid  = threadIdx.x;
  const int lane = tid & 31;
  const int wv   = tid >> 5;
  const int w0   = blockIdx.x * 32 + wv * 16;
  const int h    = blockIdx.y;
  const int b    = blockIdx.z;

  // Stage w3 fragment-major (12800 elems / 64 threads)
  for (int e = tid; e < 16 * 800; e += 64) {
    int n = e / 800, kk = e % 800;
    int t = kk >> 5, k = kk & 31;
    float v = (n < 4) ? w3[(n * 32 + k) * 25 + t] : 0.0f;
    ldsW[n][kk] = (_Float16)v;
  }
  __syncthreads();

  const int m   = lane & 15;
  const int hi  = lane >> 4;
  const int kbA = hi * 16;
  const int kbB = hi * 16;
  const _Float16* wrow = &ldsW[m][0];

  v8f acc = {};
#pragma unroll
  for (int t = 0; t < 25; ++t) {
    int dy = t / 5 - 2, dx = t % 5 - 2;
    int ih = h + dy, iw = w0 + m + dx;
    v8h alo = {}, ahi = {};
    if (ih >= 0 && ih < 256 && iw >= 0 && iw < 256) {
      const _Float16* p = h2 + (((b * 256 + ih) * 256) + iw) * 32;
      alo = *(const v8h*)(p + kbA);       // channels kbA..kbA+7   (b128)
      ahi = *(const v8h*)(p + kbA + 8);   // channels kbA+8..+15   (b128)
    }
    v8h blo = *(const v8h*)(wrow + t * 32 + kbB);
    v8h bhi = *(const v8h*)(wrow + t * 32 + kbB + 8);
    acc = __builtin_amdgcn_wmma_f32_16x16x32_f16(false, cat8(alo, ahi),
                                                 false, cat8(blo, bhi),
                                                 (short)0, acc, false, false);
  }

  if (m < 4) {
    float bias = b3[m];
#pragma unroll
    for (int r = 0; r < 8; ++r) {
      int px = w0 + r + hi * 8;
      out[(((b * 256 + h) * 256) + px) * 4 + m] = acc[r] + bias;
    }
  }
}

// ---------------------------------------------------------------------------
extern "C" void kernel_launch(void* const* d_in, const int* in_sizes, int n_in,
                              void* d_out, int out_size, void* d_ws, size_t ws_size,
                              hipStream_t stream) {
  const float* x  = (const float*)d_in[0];   // [16,32,32,4]
  const float* w1 = (const float*)d_in[1];   // [64,4,9,9]
  const float* b1 = (const float*)d_in[2];   // [64]
  const float* w2 = (const float*)d_in[3];   // [32,64,1,1]
  const float* b2 = (const float*)d_in[4];   // [32]
  const float* w3 = (const float*)d_in[5];   // [4,32,5,5]
  const float* b3 = (const float*)d_in[6];   // [4]
  float* out = (float*)d_out;

  char* ws = (char*)d_ws;
  _Float16* up = (_Float16*)ws;                              // 8 MiB
  _Float16* h2 = (_Float16*)(ws + (size_t)8 * 1024 * 1024);  // 64 MiB

  k_upsample<<<4096, 256, 0, stream>>>(x, up);
  k_conv12<<<dim3(16, 256, 16), 128, 0, stream>>>(up, w1, b1, w2, b2, h2);
  k_conv3<<<dim3(8, 256, 16), 64, 0, stream>>>(h2, w3, b3, out);
}
